// Model_28905129902405
// MI455X (gfx1250) — matrix operand order repaired and checked by reading, not compiled, not run on hardware
//
#include <hip/hip_runtime.h>
#include <hip/hip_bf16.h>
#include <math.h>

typedef __attribute__((ext_vector_type(16))) _Float16 v16h;
typedef __attribute__((ext_vector_type(8)))  _Float16 v8h;
typedef __attribute__((ext_vector_type(8)))  float    v8f;

#define T_FR  1024
#define CIN   1024
#define HID   256
#define KTOT  1040   // CIN + N_POS
#define NPOS  16
#define BATCH 64
#define CTX   32
#define NEV   16

#define XES   1064   // embed LDS column stride (halves), bank-conflict padded
#define XCS   264    // conv  LDS column stride (halves), bank-conflict padded

static __device__ __forceinline__ v16h cat8(v8h lo, v8h hi) {
  return __builtin_shufflevector(lo, hi, 0, 1, 2, 3, 4, 5, 6, 7,
                                         8, 9, 10, 11, 12, 13, 14, 15);
}
static __device__ __forceinline__ v8f wmma16(v16h a, v16h b, v8f c) {
  return __builtin_amdgcn_wmma_f32_16x16x32_f16(false, a, false, b,
                                                (short)0, c, false, false);
}

// ---------------------------------------------------------------------------
// Weight / constant conversion kernels
// ---------------------------------------------------------------------------
__global__ void cvt_f32_f16_kernel(const float* __restrict__ src,
                                   _Float16* __restrict__ dst, int n) {
  int i = blockIdx.x * blockDim.x + threadIdx.x;
  if (i < n) dst[i] = (_Float16)src[i];
}

// conv_w [l][o][c][tap] f32 -> [l][tap][o][c] f16 (contiguous A rows)
__global__ void cvt_conv_kernel(const float* __restrict__ src,
                                _Float16* __restrict__ dst) {
  int i = blockIdx.x * blockDim.x + threadIdx.x;   // 8*256*256*2
  if (i >= 8 * 256 * 256 * 2) return;
  int tap = i & 1;
  int c   = (i >> 1) & 255;
  int o   = (i >> 9) & 255;
  int l   = i >> 17;
  dst[(((size_t)(l * 2 + tap) * HID + o) * HID) + c] = (_Float16)src[i];
}

// head weights: rows 0..31 = ev_w, row 32 = sw_w, rows 33..47 = 0  (48x256)
__global__ void cvt_head_kernel(const float* __restrict__ evw,
                                const float* __restrict__ sww,
                                _Float16* __restrict__ dst) {
  int i = blockIdx.x * blockDim.x + threadIdx.x;   // 48*256
  if (i >= 48 * HID) return;
  int o = i >> 8, c = i & 255;
  float v = 0.f;
  if (o < CTX) v = evw[o * HID + c];
  else if (o == CTX) v = sww[c];
  dst[i] = (_Float16)v;
}

__global__ void pos_kernel(_Float16* __restrict__ pos) {
  int i = blockIdx.x * blockDim.x + threadIdx.x;   // 16*1024
  if (i >= NPOS * T_FR) return;
  int row = i >> 10, t = i & (T_FR - 1);
  float p   = (float)t / (float)T_FR;
  float ang = p * exp2f((float)(row & 7)) * 3.14159265358979323846f;
  pos[i] = (_Float16)((row < 8) ? sinf(ang) : cosf(ang));
}

// ---------------------------------------------------------------------------
// Embed: x[b,:,t] = W[256x1040] * cat(in,pos)[:,t] + bias
// 512 thr = 16 waves; whole 1056x16 input tile staged once to LDS (f16,
// transposed, zero-padded rows 1040..1055) -> branch-free K loop.
// ---------------------------------------------------------------------------
__global__ __launch_bounds__(512)
void embed_kernel(const float* __restrict__ xin, const _Float16* __restrict__ W,
                  const float* __restrict__ bias, const _Float16* __restrict__ pos,
                  _Float16* __restrict__ xout) {
  __shared__ _Float16 XT[16 * XES];
  const int tid  = threadIdx.x;
  const int wave = tid >> 5;
  const int lane = tid & 31;
  const int ln15 = lane & 15;
  const int b    = blockIdx.x >> 6;
  const int t0   = (blockIdx.x & 63) << 4;
  const float* xb = xin + (size_t)b * CIN * T_FR;

  for (int idx = tid; idx < 1056 * 16; idx += 512) {
    int col = idx & 15, row = idx >> 4;
    _Float16 v = (_Float16)0.f;
    if (row < CIN)       v = (_Float16)xb[(size_t)row * T_FR + t0 + col];
    else if (row < KTOT) v = pos[(row - CIN) * T_FR + t0 + col];
    XT[col * XES + row] = v;
  }
  __syncthreads();

  const int mBase = wave << 4;
  const int kSA = (lane & 16) ? 8 : 0;
  const int kSB = (lane & 16) ? 8 : 0;
  const _Float16* wr = W + (size_t)(mBase + ln15) * KTOT + kSA;
  const _Float16* x0 = XT + ln15 * XES + kSB;

  v8f acc = {};
  for (int kc = 0; kc < CIN; kc += 32) {
    v16h a  = cat8(*(const v8h*)(wr + kc), *(const v8h*)(wr + kc + 16));
    v16h bm = cat8(*(const v8h*)(x0 + kc), *(const v8h*)(x0 + kc + 16));
    acc = wmma16(a, bm, acc);
  }
  { // tail: K rows 1024..1055 (weight cols valid only 1024..1039)
    v8h z = {};
    v16h a  = cat8(*(const v8h*)(wr + CIN), z);
    v16h bm = cat8(*(const v8h*)(x0 + CIN), *(const v8h*)(x0 + CIN + 16));
    acc = wmma16(a, bm, acc);
  }
#pragma unroll
  for (int r = 0; r < 8; ++r) {
    int o = mBase + ((lane & 16) ? 8 + r : r);
    float v = acc[r] + bias[o];
    xout[((size_t)b * HID + o) * T_FR + t0 + ln15] = (_Float16)v;
  }
}

// ---------------------------------------------------------------------------
// Conv layer: y = leaky(W0*x[t] + W1*x[t+d] + b, 0.2) + x
// 512 thr = 16 waves; each wave: 16(M) x 32(N) tile, 4 WMMAs per K chunk.
// X tiles staged transposed in LDS; right padding via clamped-addr select.
// ---------------------------------------------------------------------------
__global__ __launch_bounds__(512)
void conv_layer_kernel(const _Float16* __restrict__ xin,
                       const _Float16* __restrict__ Wl,  // [2][256][256] f16
                       const float* __restrict__ bias,
                       _Float16* __restrict__ xout, int dil) {
  __shared__ _Float16 XA[32 * XCS];
  __shared__ _Float16 XB[32 * XCS];
  const int tid  = threadIdx.x;
  const int wave = tid >> 5;
  const int lane = tid & 31;
  const int ln15 = lane & 15;
  const int b    = blockIdx.x >> 5;
  const int t0   = (blockIdx.x & 31) << 5;
  const _Float16* xh = xin + (size_t)b * HID * T_FR;

  for (int idx = tid; idx < HID * 32; idx += 512) {
    int col = idx & 31, c = idx >> 5;
    XA[col * XCS + c] = xh[(size_t)c * T_FR + t0 + col];
    int tt  = t0 + col + dil;
    int ttc = (tt < T_FR) ? tt : (T_FR - 1);
    _Float16 v = xh[(size_t)c * T_FR + ttc];
    XB[col * XCS + c] = (tt < T_FR) ? v : (_Float16)0.f;
  }
  __syncthreads();

  const int mBase = wave << 4;
  const int kSA = (lane & 16) ? 8 : 0;
  const int kSB = (lane & 16) ? 8 : 0;
  const _Float16* W0 = Wl + (size_t)(mBase + ln15) * HID + kSA;
  const _Float16* W1 = W0 + (size_t)HID * HID;
  __builtin_prefetch(W1, 0, 3);
  const _Float16* pa0 = XA + ln15 * XCS + kSB;
  const _Float16* pa1 = XA + (16 + ln15) * XCS + kSB;
  const _Float16* pb0 = XB + ln15 * XCS + kSB;
  const _Float16* pb1 = XB + (16 + ln15) * XCS + kSB;

  v8f acc0 = {}, acc1 = {};
  for (int kc = 0; kc < HID; kc += 32) {
    v16h a0 = cat8(*(const v8h*)(W0 + kc),  *(const v8h*)(W0 + kc + 16));
    v16h a1 = cat8(*(const v8h*)(W1 + kc),  *(const v8h*)(W1 + kc + 16));
    v16h c0 = cat8(*(const v8h*)(pa0 + kc), *(const v8h*)(pa0 + kc + 16));
    v16h c1 = cat8(*(const v8h*)(pa1 + kc), *(const v8h*)(pa1 + kc + 16));
    v16h d0 = cat8(*(const v8h*)(pb0 + kc), *(const v8h*)(pb0 + kc + 16));
    v16h d1 = cat8(*(const v8h*)(pb1 + kc), *(const v8h*)(pb1 + kc + 16));
    acc0 = wmma16(a0, c0, acc0);
    acc0 = wmma16(a1, d0, acc0);
    acc1 = wmma16(a0, c1, acc1);
    acc1 = wmma16(a1, d1, acc1);
  }
#pragma unroll
  for (int r = 0; r < 8; ++r) {
    int o = mBase + ((lane & 16) ? 8 + r : r);
    float bia = bias[o];
    int t0a = t0 + ln15;
    int t0b = t0 + 16 + ln15;
    float v0 = acc0[r] + bia;
    v0 = (v0 > 0.f) ? v0 : 0.2f * v0;
    v0 += (float)xh[(size_t)o * T_FR + t0a];
    float v1 = acc1[r] + bia;
    v1 = (v1 > 0.f) ? v1 : 0.2f * v1;
    v1 += (float)xh[(size_t)o * T_FR + t0b];
    _Float16* orow = xout + ((size_t)b * HID + o) * T_FR;
    orow[t0a] = (_Float16)v0;
    orow[t0b] = (_Float16)v1;
  }
}

// ---------------------------------------------------------------------------
// Heads as WMMA: combined [48x256] head matrix (32 ev rows + 1 sw row + pad).
// 96 thr = 3 waves; B-fragments loaded coalesced from global f16 encoded.
// ---------------------------------------------------------------------------
__global__ __launch_bounds__(96)
void heads_kernel(const _Float16* __restrict__ enc,
                  const _Float16* __restrict__ hW,
                  const float* __restrict__ evB, const float* __restrict__ swB,
                  float* __restrict__ evecs, float* __restrict__ attn) {
  const int tid  = threadIdx.x;
  const int wave = tid >> 5;
  const int lane = tid & 31;
  const int ln15 = lane & 15;
  const int b    = blockIdx.x >> 6;
  const int t0   = (blockIdx.x & 63) << 4;
  const int mBase = wave << 4;
  const int kSA = (lane & 16) ? 8 : 0;
  const int kSB = (lane & 16) ? 8 : 0;
  const int tcol = t0 + ln15;
  const _Float16* wr = hW + (size_t)(mBase + ln15) * HID + kSA;
  const _Float16* xh = enc + (size_t)b * HID * T_FR;

  v8f acc = {};
  for (int kc = 0; kc < HID; kc += 32) {
    v16h a = cat8(*(const v8h*)(wr + kc), *(const v8h*)(wr + kc + 16));
    v16h bm;
#pragma unroll
    for (int j = 0; j < 16; ++j)
      bm[j] = xh[(size_t)(kc + kSB + (j & 7) + ((j & 8) << 1)) * T_FR + tcol];
    acc = wmma16(a, bm, acc);
  }
#pragma unroll
  for (int r = 0; r < 8; ++r) {
    int o = mBase + ((lane & 16) ? 8 + r : r);
    if (o < CTX) {
      evecs[((size_t)b * CTX + o) * T_FR + tcol] = acc[r] + evB[o];
    } else if (o == CTX) {
      float v = acc[r] + swB[0];
      attn[(size_t)b * T_FR + tcol] = (v > 0.f) ? v : 0.f;
    }
  }
}

// ---------------------------------------------------------------------------
// Top-k(16) per batch (descending, lower index wins ties), gather + scatter.
// ---------------------------------------------------------------------------
__global__ __launch_bounds__(256)
void topk_kernel(const float* __restrict__ attn, const float* __restrict__ evecs,
                 float* __restrict__ out_vecs, float* __restrict__ out_sched) {
  __shared__ float sv[T_FR];
  __shared__ float rv[256];
  __shared__ int   ri[256];
  __shared__ int   selIdx[NEV];
  __shared__ float selVal[NEV];
  const int tid = threadIdx.x;
  const int b   = blockIdx.x;

  for (int t = tid; t < T_FR; t += 256) sv[t] = attn[(size_t)b * T_FR + t];
  __syncthreads();

  for (int e = 0; e < NEV; ++e) {
    float best = -3.4e38f;
    int   bi   = T_FR;
    for (int t = tid; t < T_FR; t += 256) {
      float v = sv[t];
      if (v > best || (v == best && t < bi)) { best = v; bi = t; }
    }
    rv[tid] = best; ri[tid] = bi;
    __syncthreads();
    for (int s = 128; s > 0; s >>= 1) {
      if (tid < s) {
        float v2 = rv[tid + s]; int i2 = ri[tid + s];
        if (v2 > rv[tid] || (v2 == rv[tid] && i2 < ri[tid])) {
          rv[tid] = v2; ri[tid] = i2;
        }
      }
      __syncthreads();
    }
    if (tid == 0) {
      selIdx[e] = ri[0];
      selVal[e] = rv[0];
      sv[ri[0]] = -3.4e38f;
    }
    __syncthreads();
  }

  for (int j = tid; j < NEV * CTX; j += 256) {   // vecs gather (transpose)
    int e = j >> 5, o = j & 31;
    out_vecs[((size_t)b * NEV + e) * CTX + o] =
        evecs[((size_t)b * CTX + o) * T_FR + selIdx[e]];
  }
  float* sch = out_sched + (size_t)b * NEV * T_FR;
  for (int j = tid; j < NEV * T_FR; j += 256) sch[j] = 0.f;
  __syncthreads();
  if (tid < NEV) sch[(size_t)tid * T_FR + selIdx[tid]] = selVal[tid];
}

// ---------------------------------------------------------------------------
extern "C" void kernel_launch(void* const* d_in, const int* in_sizes, int n_in,
                              void* d_out, int out_size, void* d_ws, size_t ws_size,
                              hipStream_t stream) {
  const float* transformed = (const float*)d_in[0];
  const float* embed_w     = (const float*)d_in[1];
  const float* embed_b     = (const float*)d_in[2];
  const float* conv_w      = (const float*)d_in[3];
  const float* conv_b      = (const float*)d_in[4];
  const float* ev_w        = (const float*)d_in[5];
  const float* ev_b        = (const float*)d_in[6];
  const float* sw_w        = (const float*)d_in[7];
  const float* sw_b        = (const float*)d_in[8];

  char* ws = (char*)d_ws;
  _Float16* convW16  = (_Float16*)(ws + 0);                       // 2,097,152 B
  _Float16* embedW16 = (_Float16*)(ws + 2097152);                 //   532,480 B
  _Float16* headW16  = (_Float16*)(ws + 2629632);                 //    24,576 B
  _Float16* pos16    = (_Float16*)(ws + 2654208);                 //    32,768 B
  _Float16* xA       = (_Float16*)(ws + 2686976);                 // 33,554,432 B
  _Float16* xB       = (_Float16*)(ws + 2686976 + 33554432);      // 33,554,432 B
  float*    evecs    = (float*)(ws + 2686976 + 2u * 33554432);    //  8,388,608 B
  float*    attn     = (float*)(ws + 2686976 + 2u * 33554432 + 8388608);

  // weight / constant conversion
  cvt_conv_kernel<<<(8 * 256 * 256 * 2 + 255) / 256, 256, 0, stream>>>(conv_w, convW16);
  cvt_f32_f16_kernel<<<(HID * KTOT + 255) / 256, 256, 0, stream>>>(embed_w, embedW16, HID * KTOT);
  cvt_head_kernel<<<(48 * HID + 255) / 256, 256, 0, stream>>>(ev_w, sw_w, headW16);
  pos_kernel<<<(NPOS * T_FR + 255) / 256, 256, 0, stream>>>(pos16);

  // encoder
  embed_kernel<<<BATCH * (T_FR / 16), 512, 0, stream>>>(transformed, embedW16,
                                                        embed_b, pos16, xA);
  const int dils[8] = {1, 2, 4, 8, 16, 32, 64, 1};
  _Float16* cur = xA;
  _Float16* nxt = xB;
  for (int l = 0; l < 8; ++l) {
    conv_layer_kernel<<<BATCH * (T_FR / 32), 512, 0, stream>>>(
        cur, convW16 + (size_t)l * 2 * HID * HID, conv_b + l * HID, nxt, dils[l]);
    _Float16* tmp = cur; cur = nxt; nxt = tmp;
  }

  // heads + top-k + gather/scatter
  heads_kernel<<<BATCH * (T_FR / 16), 96, 0, stream>>>(cur, headW16, ev_b, sw_b,
                                                       evecs, attn);
  float* out_vecs  = (float*)d_out;
  float* out_sched = out_vecs + (size_t)BATCH * NEV * CTX;
  topk_kernel<<<BATCH, 256, 0, stream>>>(attn, evecs, out_vecs, out_sched);
}
